// LSTM_Model_10617159155730
// MI455X (gfx1250) — matrix operand order repaired and checked by reading, not compiled, not run on hardware
//
#include <hip/hip_runtime.h>

// Problem sizes (fixed by the reference)
#define Bsz   256
#define Tlen  1024
#define Fdim  128
#define Hdim  256
#define GATES 1024   // 4*H
#define ASTRIDE 392  // padded LDS row stride (elements): 784B -> 4-bank skew per row

typedef __attribute__((ext_vector_type(16))) __bf16 v16bf;
typedef __attribute__((ext_vector_type(8)))  __bf16 v8bf;
typedef __attribute__((ext_vector_type(4)))  __bf16 v4bf;
typedef __attribute__((ext_vector_type(8)))  float  v8f;
typedef __attribute__((ext_vector_type(4)))  float  v4f;

// Explicit global (address_space(1)) pointee types so laundered pointers
// lower to global_load_* (LOADcnt only), not flat_load_* (LOADcnt+DScnt).
typedef __bf16 __attribute__((address_space(1))) gbf;
typedef float  __attribute__((address_space(1))) gfloat;

// Scheduling-group pipeline markers (no-op if builtin unavailable).
#if defined(__has_builtin)
#if __has_builtin(__builtin_amdgcn_sched_group_barrier)
#define SGB(mask, size, id) __builtin_amdgcn_sched_group_barrier((mask), (size), (id))
#endif
#endif
#ifndef SGB
#define SGB(mask, size, id)
#endif
// masks: 0x020 = VMEM read, 0x100 = DS read, 0x008 = MFMA/WMMA

__device__ __forceinline__ __bf16 f2bf(float f) {
  unsigned u = __builtin_bit_cast(unsigned, f);
  u += 0x7FFFu + ((u >> 16) & 1u);            // round-to-nearest-even
  unsigned short s = (unsigned short)(u >> 16);
  return __builtin_bit_cast(__bf16, s);
}
__device__ __forceinline__ float bf2f(__bf16 b) {
  unsigned u = ((unsigned)__builtin_bit_cast(unsigned short, b)) << 16;
  return __builtin_bit_cast(float, u);
}
__device__ __forceinline__ float sigmoidf_fast(float x) {
  return 1.0f / (1.0f + __expf(-x));
}

// ---------------------------------------------------------------------------
// Pre-pass: transpose + convert weights to bf16 in workspace.
//   kt[n*128 + f] = kernel[f*1024 + n]      (n < 1024, f < 128)
//   rt[n*256 + k] = rec_kernel[k*1024 + n]  (n < 1024, k < 256)
// ---------------------------------------------------------------------------
__global__ __launch_bounds__(384) void prep_weights(
    const float* __restrict__ kernel, const float* __restrict__ rec,
    __bf16* __restrict__ kt, __bf16* __restrict__ rt) {
  const int n = blockIdx.x;        // 0..1023 (output column)
  const int tid = threadIdx.x;     // 0..383
  if (tid < Fdim) {
    kt[(size_t)n * Fdim + tid] = f2bf(kernel[(size_t)tid * GATES + n]);
  } else {
    const int k = tid - Fdim;      // 0..255
    rt[(size_t)n * Hdim + k] = f2bf(rec[(size_t)k * GATES + n]);
  }
}

// ---------------------------------------------------------------------------
// Persistent LSTM kernel: 16 blocks x 512 threads (16 wave32s).
// Block handles 16 batch rows for all T=1024 timesteps.
// Wave w computes z-tiles at columns {16w, 16w+256, 16w+512, 16w+768}
// (= gates i,f,g,o for h columns [16w,16w+16)), keeps cell state c in VGPRs.
// A tile in LDS is [16 x 384] bf16: K-cols 0..127 = x_t, 128..383 = h.
// ---------------------------------------------------------------------------
__global__ __launch_bounds__(512)
__attribute__((amdgpu_waves_per_eu(4, 4)))
void lstm_persist(
    const float* __restrict__ x, const float* __restrict__ bias,
    const float* __restrict__ w_out, const float* __restrict__ b_out,
    const __bf16* __restrict__ kt, const __bf16* __restrict__ rt,
    float* __restrict__ out) {
  __shared__ __bf16 a_lds[16 * ASTRIDE];

  const int tid  = threadIdx.x;
  const int w    = tid >> 5;       // wave id 0..15
  const int lane = tid & 31;
  const int lrow = lane & 15;      // A row / B column within tile
  const int lhi  = lane >> 4;      // half-select for K
  const int b0   = blockIdx.x * 16;

  // Zero initial hidden state region of A.
  for (int idx = tid; idx < 16 * Hdim; idx += 512) {
    a_lds[(idx >> 8) * ASTRIDE + Fdim + (idx & 255)] = f2bf(0.0f);
  }

  // Per-lane gate bias (column n = g*256 + 16w + lrow), loaded once.
  float bz[4];
#pragma unroll
  for (int g = 0; g < 4; ++g) bz[g] = bias[g * 256 + w * 16 + lrow];

  // Per-lane 32-bit element offsets into the transposed weight images.
  int offk[4], offr[4];
#pragma unroll
  for (int g = 0; g < 4; ++g) {
    const int n = g * 256 + w * 16 + lrow;
    offk[g] = n * Fdim + lhi * 16;   // 16 contiguous K values / lane
    offr[g] = n * Hdim + lhi * 16;
  }

  // A-operand element offsets within a 32-wide K chunk:
  //   lanes 0-15 hold K {0..15}; lanes 16-31 hold K {16..31}
  const int aoff0 = lhi * 16;
  const int aoff1 = lhi * 16 + 8;

  // x staging: wave w stages batch row (b0+w); 32 lanes x float4 = 128 floats.
  const float* xptr = x + (size_t)(b0 + w) * ((size_t)Tlen * Fdim) + (size_t)(lane * 4);
  const gfloat* xptrG = (const gfloat*)xptr;   // addrspacecast -> global loads

  // Cell state lives in registers for the whole sequence.
  v8f creg;
#pragma unroll
  for (int e = 0; e < 8; ++e) creg[e] = 0.0f;

  union Aop { v16bf v; v8bf h[2]; };
  struct BSet { v16bf b[4]; };

  for (int t = 0; t < Tlen; ++t) {
    // Launder weight bases (uniform, SGPR) each iteration so the 48 B-operand
    // loads cannot be hoisted (and spilled); re-tag as address_space(1).
    unsigned long long kb = (unsigned long long)kt;
    unsigned long long rb = (unsigned long long)rt;
    asm volatile("" : "+s"(kb));
    asm volatile("" : "+s"(rb));
    const gbf* ktG = (const gbf*)kb;
    const gbf* rtG = (const gbf*)rb;

    // B loader for combined K-chunk kc (0..3: x-kernel, 4..11: rec-kernel).
    auto loadB = [&](int kc, BSet& B) {
#pragma unroll
      for (int g = 0; g < 4; ++g) {
        const gbf* p = (kc < 4) ? (ktG + offk[g] + kc * 32)
                                : (rtG + offr[g] + (kc - 4) * 32);
        B.b[g] = *(const v16bf __attribute__((address_space(1)))*)p;
      }
    };
    auto loadA = [&](int kc, Aop& A) {
      const __bf16* arow = a_lds + lrow * ASTRIDE + kc * 32;  // K-cols contiguous
      A.h[0] = *(const v8bf*)(arow + aoff0);
      A.h[1] = *(const v8bf*)(arow + aoff1);
    };

    BSet Bbuf[3];
    Aop  Abuf[3];

    // Issue chunk-0 weight loads BEFORE the barrier (no LDS dependence):
    // their L2 latency hides behind x staging + barrier.
    loadB(0, Bbuf[0]);

    // --- stage x_t (f32 -> bf16) into A tile, coalesced ---
    {
      const v4f xv = *(const v4f __attribute__((address_space(1)))*)(xptrG + (size_t)t * Fdim);
      v4bf pk;
      pk[0] = f2bf(xv.x); pk[1] = f2bf(xv.y);
      pk[2] = f2bf(xv.z); pk[3] = f2bf(xv.w);
      *(v4bf*)&a_lds[w * ASTRIDE + (lane << 2)] = pk;
      // prefetch next timestep's x row (speculative, L2)
      const int tn = (t + 1 < Tlen) ? (t + 1) : t;
      __builtin_prefetch(xptr + (size_t)tn * Fdim, 0, 1);
    }
    __syncthreads();   // x_t staged + prev h writes visible

    // Pipeline prologue (inside compute region).
    loadA(0, Abuf[0]);
    loadB(1, Bbuf[1]); loadA(1, Abuf[1]);
    loadB(2, Bbuf[2]); loadA(2, Abuf[2]);

    // Accumulators start at the gate bias (bias[n] is constant over M).
    v8f acc[4];
#pragma unroll
    for (int g = 0; g < 4; ++g)
#pragma unroll
      for (int e = 0; e < 8; ++e) acc[g][e] = bz[g];

    // --- z = [x_t | h] @ [kernel ; rec_kernel] : K = 384 -> 12 chunks of 32,
    //     3-deep software pipeline (SSA double-buffering via slot reuse).
#pragma unroll
    for (int kc = 0; kc < 12; ++kc) {
      const int cur = kc % 3;
#pragma unroll
      for (int g = 0; g < 4; ++g) {
        acc[g] = __builtin_amdgcn_wmma_f32_16x16x32_bf16(
            false, Abuf[cur].v, false, Bbuf[cur].b[g], (short)0, acc[g], false, false);
      }
      if (kc + 3 < 12) {           // refill the slot just consumed
        loadB(kc + 3, Bbuf[cur]);
        loadA(kc + 3, Abuf[cur]);
      }
    }

    // Enforce the pipelined schedule for this region:
    //   A0 | B1 A1 | B2 A2 | (B_{kc+3} A_{kc+3} WMMAx4)... | WMMAx4 tail
    SGB(0x100, 2, 0);              // A0
    SGB(0x020, 8, 0);              // B1
    SGB(0x100, 2, 0);              // A1
    SGB(0x020, 8, 0);              // B2
    SGB(0x100, 2, 0);              // A2
#pragma unroll
    for (int kc = 0; kc < 12; ++kc) {
      if (kc + 3 < 12) {
        SGB(0x020, 8, 0);          // B(kc+3): 4 gates x 2 b128
        SGB(0x100, 2, 0);          // A(kc+3): 2 ds_load_b128
      }
      SGB(0x008, 4, 0);            // 4 WMMAs of chunk kc
    }

    __syncthreads();   // all A reads done before h is overwritten

    // --- gates + state update; write h_new (bf16) back to A tile ---
    // C layout: VGPR e, lanes 0-15 -> (M=e, N=lane); lanes 16-31 -> (M=e+8, N=lane-16)
#pragma unroll
    for (int e = 0; e < 8; ++e) {
      const float iv = sigmoidf_fast(acc[0][e]);
      const float fv = sigmoidf_fast(acc[1][e]);
      const float gv = fmaxf(acc[2][e], 0.0f);           // relu(g)
      const float ov = sigmoidf_fast(acc[3][e]);
      const float cn = fv * creg[e] + iv * gv;
      creg[e] = cn;
      const float hv = ov * fmaxf(cn, 0.0f);             // o * relu(c)
      const int M = e + lhi * 8;
      a_lds[M * ASTRIDE + Fdim + w * 16 + lrow] = f2bf(hv);
    }
  }
  __syncthreads();

  // --- out[b] = h_last[b,:] @ w_out + b_out ; wave w handles row w ---
  float partial = 0.0f;
#pragma unroll
  for (int j = lane; j < Hdim; j += 32)
    partial += bf2f(a_lds[w * ASTRIDE + Fdim + j]) * w_out[j];
#pragma unroll
  for (int off = 16; off > 0; off >>= 1)
    partial += __shfl_down(partial, off, 32);
  if (lane == 0) out[b0 + w] = partial + b_out[0];
}

// ---------------------------------------------------------------------------
extern "C" void kernel_launch(void* const* d_in, const int* in_sizes, int n_in,
                              void* d_out, int out_size, void* d_ws, size_t ws_size,
                              hipStream_t stream) {
  const float* x      = (const float*)d_in[0];  // [256,1024,128]
  const float* kernel = (const float*)d_in[1];  // [128,1024]
  const float* rec    = (const float*)d_in[2];  // [256,1024]
  const float* bias   = (const float*)d_in[3];  // [1024]
  const float* w_out  = (const float*)d_in[4];  // [256]
  const float* b_out  = (const float*)d_in[5];  // [1]
  float* out = (float*)d_out;                   // [256]

  __bf16* kt = (__bf16*)d_ws;                   // [1024,128] bf16
  __bf16* rt = kt + (size_t)GATES * Fdim;       // [1024,256] bf16  (768 KB total)

  prep_weights<<<GATES, 384, 0, stream>>>(kernel, rec, kt, rt);
  lstm_persist<<<Bsz / 16, 512, 0, stream>>>(x, bias, w_out, b_out, kt, rt, out);
}
